// LSTM_55697135894642
// MI455X (gfx1250) — matrix operand order repaired and checked by reading, not compiled, not run on hardware
//
#include <hip/hip_runtime.h>
#include <hip/hip_bf16.h>

// ---------------------------------------------------------------------------
// 2-layer LSTM over 128 traces x 32 steps (prefix-sharing collapses the 4096
// padded sequences to 128 trace chains; y[trace*32+t] = head(h2[t][trace])).
//
// Phase A: xproj = X @ W_ih^T + b_ih + b_hh  (WMMA bf16 GEMM, fully parallel),
//          stored directly in WMMA C-fragment layout for the recurrence.
// Phase B: 32-step recurrence, gates = xp + h @ W_hh^T (WMMA, W_hh in VGPRs,
//          c in VGPRs, h exchanged via a 4.3 KB LDS tile).
// Phase C: y = h2 @ W_lin^T + b_lin
// ---------------------------------------------------------------------------

#define TL     32
#define NT     128
#define DIN    64
#define HID    128
#define G4     512
#define NBATCH 4096

typedef __bf16 bf16_t;
typedef __bf16 v16bf __attribute__((ext_vector_type(16)));
typedef __bf16 v8bf  __attribute__((ext_vector_type(8)));
typedef float  v8f   __attribute__((ext_vector_type(8)));

static __device__ __forceinline__ float sigmoid_f(float x) {
    return 1.0f / (1.0f + __expf(-x));
}
static __device__ __forceinline__ float tanh_f(float x) {
    float e = __expf(2.0f * x);
    return (e - 1.0f) / (e + 1.0f);
}

// A fragment (16x32 bf16): lane m=lane&15 holds row m; chunks at off=(lane>>4)*8, off+16.
static __device__ __forceinline__ v16bf load_a_frag_f32(const float* __restrict__ A,
                                                        size_t row_base, size_t rstride,
                                                        int kb, int lane) {
    const int m   = lane & 15;
    const int off = (lane >> 4) * 8;
    const float* p = A + row_base + (size_t)m * rstride + kb + off;
    v16bf a;
#pragma unroll
    for (int i = 0; i < 8; ++i) { a[i] = (bf16_t)p[i]; a[i + 8] = (bf16_t)p[i + 16]; }
    return a;
}
static __device__ __forceinline__ v16bf load_a_frag_bf16(const bf16_t* __restrict__ A,
                                                         size_t row_base, size_t rstride,
                                                         int kb, int lane) {
    const int m   = lane & 15;
    const int off = (lane >> 4) * 8;
    const bf16_t* p = A + row_base + (size_t)m * rstride + kb + off;
    v8bf lo = *(const v8bf*)(p);
    v8bf hi = *(const v8bf*)(p + 16);
    v16bf a;
#pragma unroll
    for (int i = 0; i < 8; ++i) { a[i] = lo[i]; a[i + 8] = hi[i]; }
    return a;
}

// B fragment (32x16 bf16), B[k][n] = W[n][k], W row-major fp32 [G][ld].
// Lane holds column n=lane&15; K chunks at off=(lane>>4)*8 and off+16 (matches A fragment).
static __device__ __forceinline__ v16bf load_b_frag_f32(const float* __restrict__ W,
                                                        int col_base, int ld, int kb, int lane) {
    const int n   = lane & 15;
    const int off = (lane >> 4) * 8;
    const float* p = W + (size_t)(col_base + n) * ld + kb + off;
    v16bf b;
#pragma unroll
    for (int i = 0; i < 8; ++i) { b[i] = (bf16_t)p[i]; b[i + 8] = (bf16_t)p[i + 16]; }
    return b;
}

// A fragment from LDS bf16 h-tile [16][HS]; two aligned 16B chunks per lane.
static __device__ __forceinline__ v16bf load_a_frag_lds(const bf16_t* hb, int HS, int kb, int lane) {
    const int m   = lane & 15;
    const int off = (lane >> 4) * 8;
    const bf16_t* p = hb + m * HS + kb + off;
    v8bf lo = *(const v8bf*)(p);
    v8bf hi = *(const v8bf*)(p + 16);
    v16bf a;
#pragma unroll
    for (int i = 0; i < 8; ++i) { a[i] = lo[i]; a[i + 8] = hi[i]; }
    return a;
}

// ---------------------------------------------------------------------------
// Phase A GEMM. M-tiles are (t, 16-trace) tiles: A row index = t*TS + trace*TRS.
// Block: 4 waves, each owns one N-tile (ntile = blockIdx.y*4+wave), loops the
// 8 trace-blocks of t = blockIdx.x. Output written in C-fragment layout:
//   xp[ ((t*8 + tb)*32 + ntile)*256 + lane*8 + r ]
// so the recurrence reads it back with two b128 loads per gate tile.
// ---------------------------------------------------------------------------
template <int K, int TS, int TRS, typename T>
__global__ __launch_bounds__(128) void xproj_kernel(const T* __restrict__ A,
                                                    const float* __restrict__ W,
                                                    const float* __restrict__ b_ih,
                                                    const float* __restrict__ b_hh,
                                                    float* __restrict__ C) {
    constexpr int KSTEPS = K / 32;
    const int lane  = threadIdx.x & 31;
    const int wave  = threadIdx.x >> 5;
    const int ntile = blockIdx.y * 4 + wave;
    const int colb  = ntile * 16;
    const int t     = blockIdx.x;

    v16bf bfrag[KSTEPS];
#pragma unroll
    for (int kk = 0; kk < KSTEPS; ++kk)
        bfrag[kk] = load_b_frag_f32(W, colb, K, kk * 32, lane);

    const int   col  = colb + (lane & 15);
    const float bias = b_ih[col] + b_hh[col];

#pragma unroll
    for (int tb = 0; tb < 8; ++tb) {
        const int traceb = tb * 16 + ((lane >> 4) ? 8 : 0); // rows r..r+7 live at m = rofs+r
        // row base for m=0 of this tile (loader adds m*rstride):
        const size_t rowb    = ((size_t)t * TS + (size_t)(tb * 16) * TRS) * K;
        const size_t rstride = (size_t)TRS * K;
        v8f acc = {};
#pragma unroll
        for (int kk = 0; kk < KSTEPS; ++kk) {
            v16bf a;
            if constexpr (sizeof(T) == 4)
                a = load_a_frag_f32((const float*)A, rowb, rstride, kk * 32, lane);
            else
                a = load_a_frag_bf16((const bf16_t*)A, rowb, rstride, kk * 32, lane);
            acc = __builtin_amdgcn_wmma_f32_16x16x32_bf16(false, a, false, bfrag[kk],
                                                          (short)0, acc, false, false);
        }
        (void)traceb;
        v8f outv;
#pragma unroll
        for (int r = 0; r < 8; ++r) outv[r] = acc[r] + bias;
        float* cp = C + ((size_t)(t * 8 + tb) * 32 + ntile) * 256 + lane * 8;
        *(v8f*)cp = outv;
    }
}

// ---------------------------------------------------------------------------
// Phase B: 8 blocks (16 traces) x 8 waves. Wave w owns hidden cols [16w,16w+16)
// via interleaved gate tiles {w, w+8, w+16, w+24}: i/f/g/o share the same
// (trace,hidden) register layout -> full cell update in registers.
// W_hh resident in VGPRs for all 32 steps; h exchanged via LDS; h-stream
// written as bf16 via cooperative 16B-per-thread copies.
// ---------------------------------------------------------------------------
__global__ __launch_bounds__(256, 1) void lstm_rec_kernel(const float* __restrict__ xp,
                                                          const float* __restrict__ Whh,
                                                          bf16_t* __restrict__ hstream) {
    constexpr int HS = 136;                 // padded LDS stride (bank-friendly, 16B aligned)
    __shared__ bf16_t hbuf[16 * HS];

    const int lane   = threadIdx.x & 31;
    const int wave   = threadIdx.x >> 5;    // 0..7
    const int tb     = blockIdx.x;          // trace block
    const int traceb = tb * 16;
    const int hb     = wave * 16;           // hidden column base owned by this wave
    const int crow   = threadIdx.x >> 4;    // cooperative copy: row 0..15
    const int cchunk = threadIdx.x & 15;    // cooperative copy: 8-elem chunk 0..15

    // W_hh B-fragments resident in VGPRs for the whole recurrence (16 frags = 128 VGPRs)
    v16bf bfrag[4][4];
#pragma unroll
    for (int g = 0; g < 4; ++g)
#pragma unroll
        for (int kk = 0; kk < 4; ++kk)
            bfrag[g][kk] = load_b_frag_f32(Whh, g * HID + hb, HID, kk * 32, lane);

    for (int i = threadIdx.x; i < 16 * HS; i += blockDim.x) hbuf[i] = (bf16_t)0.0f;

    v8f c = {};                             // cell state for our (16 trace)x(16 hidden) tile

    __syncthreads();

    for (int t = 0; t < TL; ++t) {
        // gate accumulators initialized from fragment-layout xp (2x b128 per gate)
        const size_t base_t = ((size_t)t * 8 + tb) * 32 * 256;
        v8f acc[4];
#pragma unroll
        for (int g = 0; g < 4; ++g)
            acc[g] = *(const v8f*)(xp + base_t + (size_t)(g * 8 + wave) * 256 + lane * 8);
        if (t + 1 < TL)
            __builtin_prefetch(xp + base_t + (size_t)32 * 256, 0, 1);

        // recurrent GEMM: (16 x 128) @ (128 x 64-owned-cols)
#pragma unroll
        for (int kk = 0; kk < 4; ++kk) {
            v16bf a = load_a_frag_lds(hbuf, HS, kk * 32, lane);
#pragma unroll
            for (int g = 0; g < 4; ++g)
                acc[g] = __builtin_amdgcn_wmma_f32_16x16x32_bf16(false, a, false, bfrag[g][kk],
                                                                 (short)0, acc[g], false, false);
        }
        __syncthreads();                    // all reads of h_{t-1} done before overwrite

        // LSTM cell: fully in registers (i,f,g,o share this tile's layout)
        const int col  = lane & 15;
        const int rofs = (lane >> 4) * 8;
#pragma unroll
        for (int r = 0; r < 8; ++r) {
            const float gi = sigmoid_f(acc[0][r]);
            const float gf = sigmoid_f(acc[1][r]);
            const float gg = tanh_f(acc[2][r]);
            const float go = sigmoid_f(acc[3][r]);
            const float cn = gf * c[r] + gi * gg;
            c[r] = cn;
            const float h = go * tanh_f(cn);
            hbuf[(rofs + r) * HS + hb + col] = (bf16_t)h;
        }
        __syncthreads();                    // h_t visible to all waves

        // cooperative bf16 h-stream store: 16B per thread, fully coalesced
        {
            v8bf hv = *(const v8bf*)(&hbuf[crow * HS + cchunk * 8]);
            *(v8bf*)(&hstream[((size_t)t * NT + traceb + crow) * HID + cchunk * 8]) = hv;
        }
    }
}

// ---------------------------------------------------------------------------
// Phase C: y[trace*32+t] = W_lin . h2[t][trace] + b_lin
// ---------------------------------------------------------------------------
__global__ __launch_bounds__(256) void head_kernel(const bf16_t* __restrict__ h2s,
                                                   const float* __restrict__ Wl,
                                                   const float* __restrict__ bl,
                                                   float* __restrict__ y) {
    const int b = blockIdx.x * blockDim.x + threadIdx.x;
    if (b >= NBATCH) return;
    const int trace = b / TL;
    const int t     = b % TL;
    const bf16_t* h = h2s + ((size_t)t * NT + trace) * HID;
    float s = bl[0];
#pragma unroll 8
    for (int i = 0; i < HID; ++i) s += Wl[i] * (float)h[i];
    y[b] = s;
}

extern "C" void kernel_launch(void* const* d_in, const int* in_sizes, int n_in,
                              void* d_out, int out_size, void* d_ws, size_t ws_size,
                              hipStream_t stream) {
    const float* input = (const float*)d_in[0];   // [4096][64], row = trace*32 + t
    const float* Wih0  = (const float*)d_in[1];   // [512][64]
    const float* Whh0  = (const float*)d_in[2];   // [512][128]
    const float* bih0  = (const float*)d_in[3];
    const float* bhh0  = (const float*)d_in[4];
    const float* Wih1  = (const float*)d_in[5];   // [512][128]
    const float* Whh1  = (const float*)d_in[6];   // [512][128]
    const float* bih1  = (const float*)d_in[7];
    const float* bhh1  = (const float*)d_in[8];
    const float* Wlin  = (const float*)d_in[9];   // [128]
    const float* blin  = (const float*)d_in[10];
    float* y = (float*)d_out;                     // [4096]

    char* ws = (char*)d_ws;
    float*  xp = (float*)ws;                                      // 8 MB, fragment layout
    bf16_t* h1 = (bf16_t*)(ws + (size_t)NBATCH * G4 * sizeof(float));
    bf16_t* h2 = h1 + (size_t)TL * NT * HID;                      // 1 MB each

    // Layer 0 input projection: A row = t*1 + trace*32 (fp32 input)
    xproj_kernel<DIN, 1, TL, float><<<dim3(TL, 8), 128, 0, stream>>>(input, Wih0, bih0, bhh0, xp);
    // Layer 0 recurrence
    lstm_rec_kernel<<<NT / 16, 256, 0, stream>>>(xp, Whh0, h1);
    // Layer 1 input projection: A row = t*128 + trace (bf16 h1 stream)
    xproj_kernel<HID, NT, 1, bf16_t><<<dim3(TL, 8), 128, 0, stream>>>(h1, Wih1, bih1, bhh1, xp);
    // Layer 1 recurrence
    lstm_rec_kernel<<<NT / 16, 256, 0, stream>>>(xp, Whh1, h2);
    // Head
    head_kernel<<<NBATCH / 256, 256, 0, stream>>>(h2, Wlin, blin, y);
}
